// PatchEmbed_4286377361905
// MI455X (gfx1250) — matrix operand order repaired and checked by reading, not compiled, not run on hardware
//
#include <hip/hip_runtime.h>

// ---------------------------------------------------------------------------
// Octree sparse-conv stack for MI455X (gfx1250): gather + bf16 WMMA GEMM,
// batch-stat BN via atomics, ReLU, bf16 requantize between stages.
// ---------------------------------------------------------------------------

typedef unsigned int u32;
typedef __attribute__((ext_vector_type(16))) __bf16 v16bf;
typedef __attribute__((ext_vector_type(8)))  float  v8f;

struct __attribute__((aligned(16))) U4 { u32 x, y, z, w; };

union Frag {
  v16bf v;
  U4    q[2];
};

static constexpr int cN0 = 262144, cN1 = 32768, cN2 = 4096;
static constexpr int cK3 = 27, cK2 = 8, cCIN = 3, cC0 = 24, cC1 = 48, cC2 = 96;

__device__ __forceinline__ unsigned short f2bf(float f) {
  u32 u = __float_as_uint(f);
  u32 r = (u + 0x7FFFu + ((u >> 16) & 1u)) >> 16;   // round-to-nearest-even
  return (unsigned short)r;
}

// ---------------------------------------------------------------------------
// Utility kernels
// ---------------------------------------------------------------------------
__global__ void zero_kernel(float* __restrict__ p, int n) {
  int i = blockIdx.x * blockDim.x + threadIdx.x;
  if (i < n) p[i] = 0.0f;
}

__global__ void bn_stats_kernel(const float* __restrict__ sum,
                                const float* __restrict__ sumsq,
                                const float* __restrict__ gamma,
                                const float* __restrict__ beta,
                                float* __restrict__ scale,
                                float* __restrict__ shift,
                                int C, float invM) {
  int c = threadIdx.x;
  if (c < C) {
    float mu  = sum[c] * invM;
    float var = sumsq[c] * invM - mu * mu;
    float sc  = gamma[c] * rsqrtf(var + 1e-5f);
    scale[c] = sc;
    shift[c] = beta[c] - mu * sc;
  }
}

__global__ void bn_apply_bf16_kernel(const float* __restrict__ y,
                                     const float* __restrict__ scale,
                                     const float* __restrict__ shift,
                                     unsigned short* __restrict__ xb,
                                     int total, int C) {
  int i = blockIdx.x * blockDim.x + threadIdx.x;
  if (i < total) {
    int c = i % C;
    float v = fmaf(y[i], scale[c], shift[c]);
    xb[i] = f2bf(fmaxf(v, 0.0f));
  }
}

__global__ void bn_apply_f32_kernel(const float* __restrict__ y,
                                    const float* __restrict__ scale,
                                    const float* __restrict__ shift,
                                    float* __restrict__ out,
                                    int total, int C) {
  int i = blockIdx.x * blockDim.x + threadIdx.x;
  if (i < total) {
    int c = i % C;
    float v = fmaf(y[i], scale[c], shift[c]);
    out[i] = fmaxf(v, 0.0f);
  }
}

// ---------------------------------------------------------------------------
// Weight pre-pack: fp32 W[K][CIN][COUT] -> bf16 B-fragment layout.
// packB[(s*NT + nt)*512 + lane*16 + e]:
//   lanes 0-15  hold K = s*32 + (e&7) + (e>=8 ? 16 : 0)
//   lanes 16-31 hold K = s*32 + 8 + (e&7) + (e>=8 ? 16 : 0)
//   column N = nt*16 + (lane&15)
// ---------------------------------------------------------------------------
template<int K, int CIN, int COUT>
__global__ void pack_weights_kernel(const float* __restrict__ W,
                                    unsigned short* __restrict__ packB) {
  constexpr int KC    = K * CIN;
  constexpr int STEPS = (KC + 31) / 32;
  constexpr int NT    = COUT / 16;
  constexpr int TOTAL = STEPS * NT * 512;
  int idx = blockIdx.x * blockDim.x + threadIdx.x;
  if (idx >= TOTAL) return;
  const int e    = idx & 15;
  const int lane = (idx >> 4) & 31;
  const int nt   = (idx >> 9) % NT;
  const int s    = idx / (512 * NT);
  const int n    = nt * 16 + (lane & 15);
  const int kk   = s * 32 + ((lane < 16) ? 0 : 8) + (e & 7) + ((e >= 8) ? 16 : 0);
  unsigned short val = 0;
  if (kk < KC) val = f2bf(W[(size_t)kk * COUT + n]);
  packB[idx] = val;
}

// ---------------------------------------------------------------------------
// Stage A (Cin=3): VALU conv — WMMA gather granularity (8 bf16/chunk) does
// not divide Cin=3, and it's only 1 GFLOP. W lives in LDS; stats reduced in
// LDS then one atomic per channel per block.
// ---------------------------------------------------------------------------
__global__ __launch_bounds__(256)
void conv0_kernel(const float* __restrict__ data,
                  const int* __restrict__ neigh,
                  const float* __restrict__ W,
                  float* __restrict__ y,
                  float* __restrict__ sum, float* __restrict__ sumsq) {
  __shared__ float Wl[cK3 * cCIN * cC0];
  __shared__ float ssum[cC0];
  __shared__ float ssq[cC0];
  const int tid = threadIdx.x;
  for (int i = tid; i < cK3 * cCIN * cC0; i += 256) Wl[i] = W[i];
  if (tid < cC0) { ssum[tid] = 0.0f; ssq[tid] = 0.0f; }
  __syncthreads();

  const int m = blockIdx.x * 256 + tid;
  float acc[cC0];
#pragma unroll
  for (int d = 0; d < cC0; ++d) acc[d] = 0.0f;

  const int* nrow = neigh + (long long)m * cK3;
  for (int k = 0; k < cK3; ++k) {
    const int nb = nrow[k];
    const float x0 = data[nb * 3 + 0];
    const float x1 = data[nb * 3 + 1];
    const float x2 = data[nb * 3 + 2];
    const float* wk = &Wl[k * cCIN * cC0];
#pragma unroll
    for (int d = 0; d < cC0; ++d)
      acc[d] = fmaf(x0, wk[d], fmaf(x1, wk[cC0 + d], fmaf(x2, wk[2 * cC0 + d], acc[d])));
  }
#pragma unroll
  for (int d = 0; d < cC0; ++d) {
    y[(size_t)m * cC0 + d] = acc[d];
    atomicAdd(&ssum[d], acc[d]);
    atomicAdd(&ssq[d], acc[d] * acc[d]);
  }
  __syncthreads();
  if (tid < cC0) {
    atomicAdd(&sum[tid], ssum[tid]);
    atomicAdd(&sumsq[tid], ssq[tid]);
  }
}

// ---------------------------------------------------------------------------
// Stages B..E: gather + WMMA bf16 GEMM.
// One wave per 16-row tile; NTC Cout-tiles per wave (A-fragment reuse).
// A layout (16-bit A 16x32): lane<16: row=lane,   K chunks [0..7],[16..23]
//                            lane>=16: row=lane-16, K chunks [8..15],[24..31]
// Cin % 8 == 0 => each 8-elem chunk sits in one neighbor row => 1 b128 load.
// ---------------------------------------------------------------------------
template<int K, int CIN, int COUT, int NTC>
__global__ __launch_bounds__(128)
void conv_wmma_kernel(const unsigned short* __restrict__ xb,   // [Nin, CIN] bf16
                      const int* __restrict__ neigh,           // [M, K]
                      const unsigned short* __restrict__ packB,
                      float* __restrict__ y,                   // [M, COUT]
                      float* __restrict__ sum, float* __restrict__ sumsq,
                      int M) {
  constexpr int KC    = K * CIN;
  constexpr int STEPS = (KC + 31) / 32;
  constexpr int NT    = COUT / 16;

  const int lane   = threadIdx.x & 31;
  const int wv     = threadIdx.x >> 5;
  const int tile   = blockIdx.x * 4 + wv;       // 4 waves per block
  const int ntBase = blockIdx.y * NTC;
  const int half   = lane >> 4;
  const int l15    = lane & 15;
  const int row    = tile * 16 + l15;

  const int* nrow = neigh + (long long)row * K;

  v8f acc[NTC];
  const v8f vzero = {0.f, 0.f, 0.f, 0.f, 0.f, 0.f, 0.f, 0.f};
#pragma unroll
  for (int i = 0; i < NTC; ++i) acc[i] = vzero;

  for (int s = 0; s < STEPS; ++s) {
    Frag a;
    const int jbase = s * 32 + half * 8;
#pragma unroll
    for (int cpart = 0; cpart < 2; ++cpart) {
      const int j = jbase + cpart * 16;
      if (j < KC) {                      // KC % 8 == 0: chunk fully in/out
        const int k  = j / CIN;
        const int c  = j - k * CIN;      // c % 8 == 0 => 16B aligned
        const int nb = nrow[k];
        a.q[cpart] = *reinterpret_cast<const U4*>(xb + (long long)nb * CIN + c);
      } else {
        a.q[cpart] = U4{0u, 0u, 0u, 0u};
      }
    }
#pragma unroll
    for (int i = 0; i < NTC; ++i) {
      const int nt = ntBase + i;
      Frag b;
      const U4* bp = reinterpret_cast<const U4*>(
          packB + ((size_t)(s * NT + nt) * 512 + lane * 16));
      b.q[0] = bp[0];
      b.q[1] = bp[1];
      acc[i] = __builtin_amdgcn_wmma_f32_16x16x32_bf16(
          false, a.v, false, b.v, (short)0, acc[i], false, false);
    }
  }

  // D layout: VGPR r -> row = tile*16 + half*8 + r, col = nt*16 + l15
  const int rbase = tile * 16 + half * 8;
#pragma unroll
  for (int i = 0; i < NTC; ++i) {
    const int col = (ntBase + i) * 16 + l15;
    float s1 = 0.0f, s2 = 0.0f;
#pragma unroll
    for (int r = 0; r < 8; ++r) {
      float vv = acc[i][r];
      y[(size_t)(rbase + r) * COUT + col] = vv;
      s1 += vv;
      s2 += vv * vv;
    }
    atomicAdd(&sum[col], s1);
    atomicAdd(&sumsq[col], s2);
  }
  (void)M;
}

// ---------------------------------------------------------------------------
// Host side
// ---------------------------------------------------------------------------
static constexpr int kcsteps(int k, int cin) { return (k * cin + 31) / 32; }
static constexpr int packElems(int k, int cin, int cout) {
  return kcsteps(k, cin) * (cout / 16) * 512;
}

extern "C" void kernel_launch(void* const* d_in, const int* in_sizes, int n_in,
                              void* d_out, int out_size, void* d_ws, size_t ws_size,
                              hipStream_t stream) {
  const float* data   = (const float*)d_in[0];
  const int*   neigh0 = (const int*)d_in[1];
  const int*   child0 = (const int*)d_in[2];
  const int*   neigh1 = (const int*)d_in[3];
  const int*   child1 = (const int*)d_in[4];
  const int*   neigh2 = (const int*)d_in[5];
  const float* w0  = (const float*)d_in[6];
  const float* g0  = (const float*)d_in[7];
  const float* b0  = (const float*)d_in[8];
  const float* wd0 = (const float*)d_in[9];
  const float* gd0 = (const float*)d_in[10];
  const float* bd0 = (const float*)d_in[11];
  const float* w1  = (const float*)d_in[12];
  const float* g1  = (const float*)d_in[13];
  const float* b1  = (const float*)d_in[14];
  const float* wd1 = (const float*)d_in[15];
  const float* gd1 = (const float*)d_in[16];
  const float* bd1 = (const float*)d_in[17];
  const float* wp  = (const float*)d_in[18];
  const float* gp  = (const float*)d_in[19];
  const float* bp  = (const float*)d_in[20];
  (void)in_sizes; (void)n_in; (void)out_size; (void)ws_size;

  char* ws = (char*)d_ws;
  size_t off = 0;
  auto alloc = [&](size_t bytes) -> void* {
    off = (off + 255) & ~(size_t)255;
    void* p = ws + off;
    off += bytes;
    return p;
  };

  float* stats = (float*)alloc(4 * 96 * sizeof(float));
  float* sum   = stats;
  float* sumsq = stats + 96;
  float* scale = stats + 192;
  float* shift = stats + 288;

  unsigned short* pB = (unsigned short*)alloc(packElems(cK2, cC0, cC1) * 2);
  unsigned short* pC = (unsigned short*)alloc(packElems(cK3, cC1, cC1) * 2);
  unsigned short* pD = (unsigned short*)alloc(packElems(cK2, cC1, cC2) * 2);
  unsigned short* pE = (unsigned short*)alloc(packElems(cK3, cC2, cC2) * 2);

  float*          y0  = (float*)alloc((size_t)cN0 * cC0 * 4);
  unsigned short* x0b = (unsigned short*)alloc((size_t)cN0 * cC0 * 2);
  float*          y1  = (float*)alloc((size_t)cN1 * cC1 * 4);
  unsigned short* x1b = (unsigned short*)alloc((size_t)cN1 * cC1 * 2);
  float*          y2  = (float*)alloc((size_t)cN1 * cC1 * 4);
  unsigned short* x2b = (unsigned short*)alloc((size_t)cN1 * cC1 * 2);
  float*          y3  = (float*)alloc((size_t)cN2 * cC2 * 4);
  unsigned short* x3b = (unsigned short*)alloc((size_t)cN2 * cC2 * 2);
  float*          y4  = (float*)alloc((size_t)cN2 * cC2 * 4);

  // ---- weight pre-pack (every call: deterministic) ----
  pack_weights_kernel<cK2, cC0, cC1>
      <<<(packElems(cK2, cC0, cC1) + 255) / 256, 256, 0, stream>>>(wd0, pB);
  pack_weights_kernel<cK3, cC1, cC1>
      <<<(packElems(cK3, cC1, cC1) + 255) / 256, 256, 0, stream>>>(w1, pC);
  pack_weights_kernel<cK2, cC1, cC2>
      <<<(packElems(cK2, cC1, cC2) + 255) / 256, 256, 0, stream>>>(wd1, pD);
  pack_weights_kernel<cK3, cC2, cC2>
      <<<(packElems(cK3, cC2, cC2) + 255) / 256, 256, 0, stream>>>(wp, pE);

  // ---- Stage A: conv3 (Cin=3) on VALU ----
  zero_kernel<<<1, 192, 0, stream>>>(stats, 192);
  conv0_kernel<<<cN0 / 256, 256, 0, stream>>>(data, neigh0, w0, y0, sum, sumsq);
  bn_stats_kernel<<<1, 96, 0, stream>>>(sum, sumsq, g0, b0, scale, shift,
                                        cC0, 1.0f / cN0);
  bn_apply_bf16_kernel<<<(cN0 * cC0 + 255) / 256, 256, 0, stream>>>(
      y0, scale, shift, x0b, cN0 * cC0, cC0);

  // ---- Stage B: downsample conv (K=8, 24->48), WMMA ----
  zero_kernel<<<1, 192, 0, stream>>>(stats, 192);
  conv_wmma_kernel<cK2, cC0, cC1, 3>
      <<<dim3(cN1 / 64, 1), 128, 0, stream>>>(x0b, child0, pB, y1, sum, sumsq, cN1);
  bn_stats_kernel<<<1, 96, 0, stream>>>(sum, sumsq, gd0, bd0, scale, shift,
                                        cC1, 1.0f / cN1);
  bn_apply_bf16_kernel<<<(cN1 * cC1 + 255) / 256, 256, 0, stream>>>(
      y1, scale, shift, x1b, cN1 * cC1, cC1);

  // ---- Stage C: conv3 (K=27, 48->48), WMMA ----
  zero_kernel<<<1, 192, 0, stream>>>(stats, 192);
  conv_wmma_kernel<cK3, cC1, cC1, 3>
      <<<dim3(cN1 / 64, 1), 128, 0, stream>>>(x1b, neigh1, pC, y2, sum, sumsq, cN1);
  bn_stats_kernel<<<1, 96, 0, stream>>>(sum, sumsq, g1, b1, scale, shift,
                                        cC1, 1.0f / cN1);
  bn_apply_bf16_kernel<<<(cN1 * cC1 + 255) / 256, 256, 0, stream>>>(
      y2, scale, shift, x2b, cN1 * cC1, cC1);

  // ---- Stage D: downsample conv (K=8, 48->96), WMMA ----
  zero_kernel<<<1, 192, 0, stream>>>(stats, 192);
  conv_wmma_kernel<cK2, cC1, cC2, 3>
      <<<dim3(cN2 / 64, 2), 128, 0, stream>>>(x2b, child1, pD, y3, sum, sumsq, cN2);
  bn_stats_kernel<<<1, 96, 0, stream>>>(sum, sumsq, gd1, bd1, scale, shift,
                                        cC2, 1.0f / cN2);
  bn_apply_bf16_kernel<<<(cN2 * cC2 + 255) / 256, 256, 0, stream>>>(
      y3, scale, shift, x3b, cN2 * cC2, cC2);

  // ---- Stage E: proj conv (K=27, 96->96), WMMA, fp32 out ----
  zero_kernel<<<1, 192, 0, stream>>>(stats, 192);
  conv_wmma_kernel<cK3, cC2, cC2, 3>
      <<<dim3(cN2 / 64, 2), 128, 0, stream>>>(x3b, neigh2, pE, y4, sum, sumsq, cN2);
  bn_stats_kernel<<<1, 96, 0, stream>>>(sum, sumsq, gp, bp, scale, shift,
                                        cC2, 1.0f / cN2);
  bn_apply_f32_kernel<<<(cN2 * cC2 + 255) / 256, 256, 0, stream>>>(
      y4, scale, shift, (float*)d_out, cN2 * cC2, cC2);
}
